// FutureDecoder_33174327394637
// MI455X (gfx1250) — matrix operand order repaired and checked by reading, not compiled, not run on hardware
//
#include <hip/hip_runtime.h>
#include <stdint.h>

// ---------------------------------------------------------------------------
// GRU trajectory decoder for gfx1250 (MI455X).
// g^T = W * h^T with v_wmma_f32_16x16x32_bf16. W_hh/W_out columns are stored in a
// bit3<->bit4 swapped ordering so the WMMA D-layout registers map directly to
// next step's B-operand K-slots (no cross-lane exchange). Weights are bf16 in
// LDS (W_hh / W_out columns pre-permuted at load time); hidden state lives in
// VGPRs as f32 accumulator tiles. LICM of per-step LDS loads is defeated with
// an opaque zero offset (asm volatile) so nothing is hoisted+spilled; loads
// stay ds_load_b128 with automatic waitcnt. The 72 gate WMMAs are software-
// pipelined (depth 1) via a constexpr descriptor table: the A-fragment for
// WMMA s+1 is loaded before WMMA s issues, so the steady state is
// load(s+1) / wmma(s) instead of load/wait0/wmma.
// ---------------------------------------------------------------------------

typedef __attribute__((ext_vector_type(16))) __bf16    bf16x16;
typedef __attribute__((ext_vector_type(2)))  __bf16    bf16x2;
typedef __attribute__((ext_vector_type(8)))  float     v8f;
typedef __attribute__((ext_vector_type(4)))  float     f32x4;
typedef __attribute__((ext_vector_type(4)))  uint32_t  u32x4;

union FragU { uint32_t u[8]; u32x4 q[2]; bf16x16 v; };
union V8U   { float f[8]; f32x4 q[2]; v8f v; };

#define HID 96
#define EMB 32
#define G3  288   // 3*HID gate rows
#define SEQ 48

// swap bits 3 and 4 of a hidden index (valid within each 32-block of 0..95)
__device__ __forceinline__ int sw34(int q) {
  return (q & ~0x18) | ((q & 8) << 1) | ((q & 16) >> 1);
}

__device__ __forceinline__ uint32_t pkbf(float a, float b) {
#if __has_builtin(__builtin_amdgcn_cvt_pk_bf16_f32)
  bf16x2 t = __builtin_amdgcn_cvt_pk_bf16_f32(a, b);
  return __builtin_bit_cast(uint32_t, t);
#else
  // round-to-nearest-even f32 -> bf16, packed pair (a lo, b hi)
  uint32_t ua = __float_as_uint(a), ub = __float_as_uint(b);
  ua = (ua + 0x7fffu + ((ua >> 16) & 1u)) >> 16;
  ub = (ub + 0x7fffu + ((ub >> 16) & 1u)) & 0xffff0000u;
  return (ua & 0xffffu) | ub;
#endif
}

__device__ __forceinline__ float sigm(float x) {
  return __builtin_amdgcn_rcpf(1.f + __expf(-x));
}
__device__ __forceinline__ float tanhfx(float x) {
  return fmaf(2.f, __builtin_amdgcn_rcpf(1.f + __expf(-2.f * x)), -1.f);
}

__device__ __forceinline__ v8f wmma_bf16(bf16x16 a, bf16x16 b, v8f c) {
  return __builtin_amdgcn_wmma_f32_16x16x32_bf16(false, a, false, b, (short)0, c, false, false);
}

// A-fragment (16xK tile) from LDS bf16, row-major (plain loads -> ds_load_b128).
__device__ __forceinline__ bf16x16 ldA(const uint16_t* w, int stride, int rowBase,
                                       int k0, int lane) {
  const int r = rowBase + (lane & 15);
  const int k = k0 + ((lane & 16) ? 16 : 0);
  FragU f;
  f.q[0] = *(const u32x4*)(w + r * stride + k);
  f.q[1] = *(const u32x4*)(w + r * stride + k + 8);
  return f.v;
}

// 8 consecutive f32 from LDS -> v8f (C-operand bias rows).
__device__ __forceinline__ v8f ldB8(const float* b, int idx) {
  V8U r;
  r.q[0] = *(const f32x4*)(b + idx);
  r.q[1] = *(const f32x4*)(b + idx + 4);
  return r.v;
}

// Descriptor for one gate WMMA: which accumulator, which weight matrix,
// which gate block (row offset multiple of HID), K chunk, and B operand.
struct AStep { uint8_t accIdx, isIh, rowBlk, k0, bsel; };
static constexpr AStep aseq[12] = {
  {0, 0, 0, 0,  0}, {1, 0, 1, 0,  0}, {2, 0, 2, 0,  0},   // W_hh chunk 0
  {0, 0, 0, 32, 1}, {1, 0, 1, 32, 1}, {2, 0, 2, 32, 1},   // W_hh chunk 1
  {0, 0, 0, 64, 2}, {1, 0, 1, 64, 2}, {2, 0, 2, 64, 2},   // W_hh chunk 2
  {0, 1, 0, 0,  3}, {1, 1, 1, 0,  3}, {3, 1, 2, 0,  3},   // W_ih (x path)
};

__global__ __launch_bounds__(256) void gru_decoder(
    const float* __restrict__ last_pos, const float* __restrict__ past,
    const float* __restrict__ fut,
    const float* __restrict__ Wemb, const float* __restrict__ bemb,
    const float* __restrict__ Wih,  const float* __restrict__ Whh,
    const float* __restrict__ bih,  const float* __restrict__ bhh,
    const float* __restrict__ Wout, const float* __restrict__ bout,
    float* __restrict__ out, int B)
{
  __shared__ __align__(16) uint16_t sWhh[G3 * HID];   // bf16, cols sw34-permuted
  __shared__ __align__(16) uint16_t sWih[G3 * EMB];   // bf16, unpermuted
  __shared__ __align__(16) uint16_t sWoutP[16 * HID]; // bf16, rows 3..15 zero, cols sw34
  __shared__ __align__(16) float    sBrz[2 * HID];    // b_ih + b_hh (r,z rows)
  __shared__ __align__(16) float    sBhn[HID];        // b_hh n-rows
  __shared__ __align__(16) float    sBxn[HID];        // b_ih n-rows
  __shared__ __align__(16) f32x4    sWemb[EMB];       // (W_emb[e][0..2], b_emb[e])

  const int tid = threadIdx.x;

  // ---- cooperative weight preload: f32->bf16, column permutation for W_hh/W_out
  for (int i = tid; i < G3 * (HID / 2); i += 256) {   // 288 rows x 48 col-pairs
    const int r = i / 48, kp = i % 48;
    const int c0 = sw34(2 * kp);                      // sw34 keeps pairs adjacent
    ((uint32_t*)sWhh)[r * 48 + kp] = pkbf(Whh[r * HID + c0], Whh[r * HID + c0 + 1]);
  }
  for (int i = tid; i < G3 * (EMB / 2); i += 256)
    ((uint32_t*)sWih)[i] = pkbf(Wih[2 * i], Wih[2 * i + 1]);
  for (int i = tid; i < 16 * (HID / 2); i += 256) {   // 16 rows x 48 col-pairs
    const int r = i / 48, kp = i % 48;
    const int c0 = sw34(2 * kp);
    ((uint32_t*)sWoutP)[r * 48 + kp] =
        (r < 3) ? pkbf(Wout[r * HID + c0], Wout[r * HID + c0 + 1]) : 0u;
  }
  for (int i = tid; i < 2 * HID; i += 256) sBrz[i] = bih[i] + bhh[i];
  for (int i = tid; i < HID; i += 256) { sBhn[i] = bhh[2 * HID + i]; sBxn[i] = bih[2 * HID + i]; }
  for (int i = tid; i < EMB; i += 256) {
    f32x4 t = {Wemb[3 * i], Wemb[3 * i + 1], Wemb[3 * i + 2], bemb[i]};
    sWemb[i] = t;
  }
  __syncthreads();

  const int  lane  = tid & 31;
  const int  wave  = tid >> 5;
  const int  hoff  = (lane & 16) ? 8 : 0;  // M-row offset of upper lane half
  const bool loh   = (lane < 16);
  const int  batch = blockIdx.x * 128 + wave * 16 + (lane & 15);

  // ---- init hidden state, D layout, identity storage order:
  // hD[j][v] = h0[16j + v + hoff] for this lane's batch column.
  v8f hD[6];
  #pragma unroll
  for (int j = 0; j < 6; ++j) {
    const float* src = (j < 3) ? (past + (size_t)batch * 48 + 16 * j + hoff)
                               : (fut  + (size_t)batch * 48 + 16 * (j - 3) + hoff);
    f32x4 lo = *(const f32x4*)src;
    f32x4 hi = *(const f32x4*)(src + 4);
    v8f t = {lo.x, lo.y, lo.z, lo.w, hi.x, hi.y, hi.z, hi.w};
    hD[j] = t;
  }

  float rel0 = last_pos[(size_t)batch * 3 + 0];
  float rel1 = last_pos[(size_t)batch * 3 + 1];
  float rel2 = last_pos[(size_t)batch * 3 + 2];
  const float bo0 = bout[0], bo1 = bout[1], bo2 = bout[2];
  const int ebase = loh ? 0 : 16;  // embedding K-range owned by this lane half

  // B operands: Ball[0..2] = h chunks, Ball[3] = x. Thanks to the sw34 weight
  // column permutation, D-layout regs map directly onto B K-slots (chunk c =
  // tiles {2c, 2c+1}, same packing for both lane halves, no lane exchange).
  FragU Ball[4];
  #pragma unroll
  for (int c = 0; c < 3; ++c)
    #pragma unroll
    for (int r = 0; r < 4; ++r) {
      Ball[c].u[r]     = pkbf(hD[2 * c][2 * r],     hD[2 * c][2 * r + 1]);
      Ball[c].u[4 + r] = pkbf(hD[2 * c + 1][2 * r], hD[2 * c + 1][2 * r + 1]);
    }

  #pragma unroll 1
  for (int step = 0; step < SEQ; ++step) {
    // Opaque zero: regenerated each iteration so LDS addresses are loop-
    // variant -> no LICM / scratch spills; loads stay ds_load_b128.
    int zoff = 0;
    asm volatile("" : "+v"(zoff));
    const uint16_t* whh   = sWhh + zoff;
    const uint16_t* wih   = sWih + zoff;
    const uint16_t* woutp = sWoutP + zoff;
    const float*    brz   = sBrz + zoff;
    const float*    bhn   = sBhn + zoff;
    const float*    bxn   = sBxn + zoff;
    const f32x4*    wemb  = sWemb + zoff;

    // ---- B fragment for x-path: x = rel @ Wemb^T + b_emb (32x16, lane-local)
    #pragma unroll
    for (int r = 0; r < 8; ++r) {
      f32x4 w0 = wemb[ebase + 2 * r];
      f32x4 w1 = wemb[ebase + 2 * r + 1];
      float x0 = fmaf(rel0, w0.x, fmaf(rel1, w0.y, fmaf(rel2, w0.z, w0.w)));
      float x1 = fmaf(rel0, w1.x, fmaf(rel1, w1.y, fmaf(rel2, w1.z, w1.w)));
      Ball[3].u[r] = pkbf(x0, x1);
    }

    // ---- gate matmuls, software-pipelined A-fragment stream (depth 1) ----
    bf16x16 aPre = ldA(whh, HID, 0, 0, lane);  // (j=0, s=0)
    #pragma unroll
    for (int j = 0; j < 6; ++j) {
      const int row = 16 * j;
      v8f acc[4];
      acc[0] = ldB8(brz,       row + hoff);    // biases as WMMA C operands
      acc[1] = ldB8(brz + HID, row + hoff);
      acc[2] = ldB8(bhn,       row + hoff);
      acc[3] = ldB8(bxn,       row + hoff);

      #pragma unroll
      for (int s = 0; s < 12; ++s) {
        const bf16x16 aCur = aPre;
        // prefetch the A fragment for the next WMMA (crossing j boundaries)
        const int ns = (s + 1) % 12;
        const int nj = (s == 11) ? j + 1 : j;
        if (nj < 6) {
          aPre = aseq[ns].isIh
                     ? ldA(wih, EMB, aseq[ns].rowBlk * HID + 16 * nj, aseq[ns].k0, lane)
                     : ldA(whh, HID, aseq[ns].rowBlk * HID + 16 * nj, aseq[ns].k0, lane);
        }
        acc[aseq[s].accIdx] = wmma_bf16(aCur, Ball[aseq[s].bsel].v, acc[aseq[s].accIdx]);
      }

      // gates + state update (f32, in D layout)
      v8f hn;
      #pragma unroll
      for (int v = 0; v < 8; ++v) {
        float r = sigm(acc[0][v]);
        float z = sigm(acc[1][v]);
        float n = tanhfx(acc[3][v] + r * acc[2][v]);
        hn[v] = (1.f - z) * n + z * hD[j][v];
      }
      hD[j] = hn;
    }

    // ---- output projection A fragments (issued before the Bh rebuild so the
    //      pack VALU below hides their LDS latency)
    const bf16x16 aO0 = ldA(woutp, HID, 0, 0,  lane);
    const bf16x16 aO1 = ldA(woutp, HID, 0, 32, lane);
    const bf16x16 aO2 = ldA(woutp, HID, 0, 64, lane);

    // ---- rebuild h B-fragments from h_new (lane-local packing only)
    #pragma unroll
    for (int c = 0; c < 3; ++c)
      #pragma unroll
      for (int r = 0; r < 4; ++r) {
        Ball[c].u[r]     = pkbf(hD[2 * c][2 * r],     hD[2 * c][2 * r + 1]);
        Ball[c].u[4 + r] = pkbf(hD[2 * c + 1][2 * r], hD[2 * c + 1][2 * r + 1]);
      }

    // ---- rel = h_new @ W_out^T via 3 WMMAs (rows 3..15 zero-padded)
    v8f cO = {0.f, 0.f, 0.f, 0.f, 0.f, 0.f, 0.f, 0.f};
    cO = wmma_bf16(aO0, Ball[0].v, cO);
    cO = wmma_bf16(aO1, Ball[1].v, cO);
    cO = wmma_bf16(aO2, Ball[2].v, cO);
    // rows 0..2 live in lanes<16 (upper-half rows are the zero padding), so a
    // single xor-16 add gives every lane the full value.
    rel0 = cO[0] + __shfl_xor(cO[0], 16, 32) + bo0;
    rel1 = cO[1] + __shfl_xor(cO[1], 16, 32) + bo1;
    rel2 = cO[2] + __shfl_xor(cO[2], 16, 32) + bo2;

    if (loh) {
      float* op = out + ((size_t)step * B + batch) * 3;
      op[0] = rel0; op[1] = rel1; op[2] = rel2;
    }
  }
}

extern "C" void kernel_launch(void* const* d_in, const int* in_sizes, int n_in,
                              void* d_out, int out_size, void* d_ws, size_t ws_size,
                              hipStream_t stream) {
  const float* last = (const float*)d_in[0];
  const float* past = (const float*)d_in[1];
  const float* fut  = (const float*)d_in[2];
  const float* Wemb = (const float*)d_in[3];
  const float* bemb = (const float*)d_in[4];
  const float* Wih  = (const float*)d_in[5];
  const float* Whh  = (const float*)d_in[6];
  const float* bih  = (const float*)d_in[7];
  const float* bhh  = (const float*)d_in[8];
  const float* Wout = (const float*)d_in[9];
  const float* bout = (const float*)d_in[10];
  const int B = in_sizes[0] / 3;           // 131072
  dim3 grid(B / 128);                      // 8 waves * 16 batch rows per block
  gru_decoder<<<grid, 256, 0, stream>>>(last, past, fut, Wemb, bemb, Wih, Whh,
                                        bih, bhh, Wout, bout, (float*)d_out, B);
}
